// GNN_56942676410827
// MI455X (gfx1250) — matrix operand order repaired and checked by reading, not compiled, not run on hardware
//
#include <hip/hip_runtime.h>
#include <hip/hip_bf16.h>

typedef __attribute__((ext_vector_type(16))) _Float16 v16h;
typedef __attribute__((ext_vector_type(8)))  _Float16 v8h;
typedef __attribute__((ext_vector_type(8)))  float    v8f;

#define WAVES_PER_BLOCK 8
#define TPB (WAVES_PER_BLOCK * 32)

// ---------------- WMMA tile helpers (wave32, gfx1250 layouts) ----------------
// A fragment: 16x32 f16 tile from LDS, row stride `stride` halfs.
// lanes 0-15: M=lane,   K = kbase + {0..7, 16..23}
// lanes 16-31:M=lane-16,K = kbase + {8..15, 24..31}
__device__ inline v16h lds_load_A(const _Float16* act, int kChunk, int stride, int lane) {
  const int row = lane & 15;
  const int kb  = kChunk * 32 + ((lane & 16) ? 8 : 0);
  const _Float16* p = act + row * stride + kb;
  v8h lo = *(const v8h*)p;          // K = kb .. kb+7
  v8h hi = *(const v8h*)(p + 16);   // K = kb+16 .. kb+23
  return __builtin_shufflevector(lo, hi, 0,1,2,3,4,5,6,7,8,9,10,11,12,13,14,15);
}

// B fragment: 32x16 f16. Weights stored transposed in LDS: wt[n*Kpad + k].
// lanes 0-15 hold K={0..7,16..23} of column N=lane; lanes 16-31 hold K={8..15,24..31} of N=lane-16.
__device__ inline v16h lds_load_B(const _Float16* wt, int nTile, int kChunk, int Kpad, int lane) {
  const int n  = nTile * 16 + (lane & 15);
  const int kb = kChunk * 32 + ((lane & 16) ? 8 : 0);
  const _Float16* p = wt + n * Kpad + kb;
  v8h lo = *(const v8h*)p;
  v8h hi = *(const v8h*)(p + 16);
  return __builtin_shufflevector(lo, hi, 0,1,2,3,4,5,6,7,8,9,10,11,12,13,14,15);
}

// D layout: VGPR r -> M = r + ((lane&16)?8:0), N = lane&15 (+16*nTile)
__device__ inline void store_D(_Float16* out, v8f d, const float* __restrict__ bias,
                               int nTile, int Npad, int nValid, int lane, bool relu) {
  const int n    = nTile * 16 + (lane & 15);
  const int moff = (lane & 16) ? 8 : 0;
  const float bv = (n < nValid) ? bias[n] : 0.f;
#pragma unroll
  for (int r = 0; r < 8; ++r) {
    float v = d[r] + bv;
    if (relu) v = fmaxf(v, 0.f);
    out[(r + moff) * Npad + n] = (_Float16)((n < nValid) ? v : 0.f);
  }
}

__device__ inline void mlp_layer(const _Float16* actIn, _Float16* actOut,
                                 const _Float16* wt, const float* __restrict__ bias,
                                 int Kpad, int Npad, int nValid, int lane, bool relu) {
  const int nTiles = Npad >> 4, kChunks = Kpad >> 5;
  for (int nt = 0; nt < nTiles; ++nt) {
    v8f acc = {};
    for (int kc = 0; kc < kChunks; ++kc) {
      v16h a = lds_load_A(actIn, kc, Kpad, lane);
      v16h b = lds_load_B(wt, nt, kc, Kpad, lane);
      acc = __builtin_amdgcn_wmma_f32_16x16x32_f16(false, a, false, b, (short)0, acc, false, false);
    }
    store_D(actOut, acc, bias, nt, Npad, nValid, lane, relu);
  }
}

// Final layer of an MLP: single 16-wide output tile kept in registers.
__device__ inline v8f mlp_layer_acc(const _Float16* actIn, const _Float16* wt, int Kpad, int lane) {
  const int kChunks = Kpad >> 5;
  v8f acc = {};
  for (int kc = 0; kc < kChunks; ++kc) {
    v16h a = lds_load_A(actIn, kc, Kpad, lane);
    v16h b = lds_load_B(wt, 0, kc, Kpad, lane);
    acc = __builtin_amdgcn_wmma_f32_16x16x32_f16(false, a, false, b, (short)0, acc, false, false);
  }
  return acc;
}

// Async global->LDS DMA staging of weight panels (CDNA5 GLOBAL_LOAD_ASYNC_TO_LDS_B128,
// tracked by ASYNCcnt). Each lane moves 16B per issue; no VGPR round trip.
__device__ inline void copy_lds_async(const _Float16* __restrict__ g, _Float16* s,
                                      int nHalf, int tid) {
  const int nv = nHalf >> 3;  // 16B packets
  for (int i = tid; i < nv; i += TPB) {
    unsigned lds_off = (unsigned)(size_t)(s + (size_t)i * 8);
    unsigned long long gaddr = (unsigned long long)(g + (size_t)i * 8);
    asm volatile("global_load_async_to_lds_b128 %0, %1, off"
                 :: "v"(lds_off), "v"(gaddr) : "memory");
  }
}

__device__ inline void async_wait_all() {
  asm volatile("s_wait_asynccnt 0" ::: "memory");
}

// ---------------- padded dims ----------------
// message MLP: 35 -> 70 -> 140 -> 20 -> 16
#define M1K 64
#define M1N 96
#define M2K 96
#define M2N 160
#define M3K 160
#define M3N 32
#define M4K 32
// update MLP: 32 -> 70 -> 16
#define U1K 32
#define U1N 96
#define U2K 96
// final MLP: 16 -> 64 -> 32 -> 3
#define F1K 32
#define F1N 64
#define F2K 64
#define F2N 32
#define F3K 32

// ---------------- small utility kernels ----------------
__global__ void zero_kernel(float* p, int n) {
  int i = blockIdx.x * blockDim.x + threadIdx.x;
  if (i < n) p[i] = 0.f;
}

__global__ void prep_w_kernel(const float* __restrict__ W, _Float16* __restrict__ wt,
                              int din, int dout, int Kpad, int Npad) {
  int i = blockIdx.x * blockDim.x + threadIdx.x;
  if (i >= Npad * Kpad) return;
  int n = i / Kpad, k = i - n * Kpad;
  float v = (n < dout && k < din) ? W[(size_t)k * dout + n] : 0.f;
  wt[i] = (_Float16)v;  // transpose + pad: wt[n*Kpad + k] = W[k][n]
}

__global__ void degree_kernel(const int* __restrict__ dstIdx, float* deg, int E) {
  int e = blockIdx.x * blockDim.x + threadIdx.x;
  if (e < E) atomicAdd(&deg[dstIdx[e]], 1.0f);
}

__global__ void count_kernel(const int* __restrict__ batch, float* cnt, int N) {
  int i = blockIdx.x * blockDim.x + threadIdx.x;
  if (i < N) atomicAdd(&cnt[batch[i]], 1.0f);
}

__global__ void encoder_kernel(const float* __restrict__ x, const float* __restrict__ W,
                               const float* __restrict__ b, float* __restrict__ h, int N) {
  int i = blockIdx.x * blockDim.x + threadIdx.x;
  if (i >= N * 16) return;
  int n = i >> 4, c = i & 15;
  const float* xr = x + (size_t)n * 16;
  float acc = b[c];
#pragma unroll
  for (int k = 0; k < 16; ++k) acc += xr[k] * W[k * 16 + c];
  h[i] = acc;
}

// ---------------- edge message kernel ----------------
__global__ __launch_bounds__(TPB) void edge_msg_kernel(
    const float* __restrict__ h, const int* __restrict__ srcIdx, const int* __restrict__ dstIdx,
    const float* __restrict__ eattr,
    const _Float16* __restrict__ gw1, const _Float16* __restrict__ gw2,
    const _Float16* __restrict__ gw3, const _Float16* __restrict__ gw4,
    const float* __restrict__ b1, const float* __restrict__ b2,
    const float* __restrict__ b3, const float* __restrict__ b4,
    float* __restrict__ aggr, int E)
{
  __shared__ __align__(16) _Float16 sW1[M1N * M1K];
  __shared__ __align__(16) _Float16 sW2[M2N * M2K];
  __shared__ __align__(16) _Float16 sW3[M3N * M3K];
  __shared__ __align__(16) _Float16 sW4[16 * M4K];
  __shared__ __align__(16) _Float16 sBufA[WAVES_PER_BLOCK][16 * M2K];  // 16x96 max
  __shared__ __align__(16) _Float16 sBufB[WAVES_PER_BLOCK][16 * M3K];  // 16x160 max

  const int tid = threadIdx.x, lane = tid & 31, wave = tid >> 5;
  copy_lds_async(gw1, sW1, M1N * M1K, tid);
  copy_lds_async(gw2, sW2, M2N * M2K, tid);
  copy_lds_async(gw3, sW3, M3N * M3K, tid);
  copy_lds_async(gw4, sW4, 16 * M4K, tid);
  async_wait_all();
  __syncthreads();

  _Float16* bufA = sBufA[wave];
  _Float16* bufB = sBufB[wave];

  const int nTiles = (E + 15) >> 4;
  const int waveGid = blockIdx.x * WAVES_PER_BLOCK + wave;
  const int nWaves  = gridDim.x * WAVES_PER_BLOCK;

  for (int t = waveGid; t < nTiles; t += nWaves) {
    const int eBase = t << 4;
    // gather: msg_in = [h[dst] | h[src] | edge_attr | 0-pad] as 16x64 f16 in bufB
    {
      const int r = lane & 15;
      const int e = eBase + r;
      const bool valid = e < E;
      _Float16* row = bufB + r * M1K;
      const int node = valid ? ((lane < 16) ? dstIdx[e] : srcIdx[e]) : 0;
      const float4* hp = (const float4*)(h + (size_t)node * 16);
      const int cb = (lane < 16) ? 0 : 16;
#pragma unroll
      for (int j = 0; j < 4; ++j) {
        float4 v = hp[j];
        if (!valid) { v.x = 0.f; v.y = 0.f; v.z = 0.f; v.w = 0.f; }
        row[cb + j * 4 + 0] = (_Float16)v.x;
        row[cb + j * 4 + 1] = (_Float16)v.y;
        row[cb + j * 4 + 2] = (_Float16)v.z;
        row[cb + j * 4 + 3] = (_Float16)v.w;
      }
      if (lane >= 16) {
#pragma unroll
        for (int c = 0; c < 3; ++c)
          row[32 + c] = (_Float16)(valid ? eattr[(size_t)e * 3 + c] : 0.f);
#pragma unroll
        for (int c = 35; c < M1K; ++c) row[c] = (_Float16)0.f;
      }
    }
    mlp_layer(bufB, bufA, sW1, b1, M1K, M1N, 70,  lane, true);
    mlp_layer(bufA, bufB, sW2, b2, M2K, M2N, 140, lane, true);
    mlp_layer(bufB, bufA, sW3, b3, M3K, M3N, 20,  lane, true);
    v8f acc = mlp_layer_acc(bufA, sW4, M4K, lane);
    // scatter-add message into aggr[dst]
    {
      const int n = lane & 15;
      const int moff = (lane & 16) ? 8 : 0;
      const float bv = b4[n];
#pragma unroll
      for (int r = 0; r < 8; ++r) {
        const int e = eBase + r + moff;
        if (e < E) {
          const int dn = dstIdx[e];
          atomicAdd(&aggr[(size_t)dn * 16 + n], acc[r] + bv);
        }
      }
    }
  }
}

// ---------------- node update kernel ----------------
__global__ __launch_bounds__(TPB) void node_update_kernel(
    const float* __restrict__ hIn, const float* __restrict__ aggr, const float* __restrict__ deg,
    const _Float16* __restrict__ gw1, const _Float16* __restrict__ gw2,
    const float* __restrict__ b1, const float* __restrict__ b2,
    float* __restrict__ hOut, int N)
{
  __shared__ __align__(16) _Float16 sW1[U1N * U1K];
  __shared__ __align__(16) _Float16 sW2[16 * U2K];
  __shared__ __align__(16) _Float16 sBuf0[WAVES_PER_BLOCK][16 * U1K];
  __shared__ __align__(16) _Float16 sBuf1[WAVES_PER_BLOCK][16 * U2K];

  const int tid = threadIdx.x, lane = tid & 31, wave = tid >> 5;
  copy_lds_async(gw1, sW1, U1N * U1K, tid);
  copy_lds_async(gw2, sW2, 16 * U2K, tid);
  async_wait_all();
  __syncthreads();

  _Float16* buf0 = sBuf0[wave];
  _Float16* buf1 = sBuf1[wave];

  const int nTiles = (N + 15) >> 4;
  const int waveGid = blockIdx.x * WAVES_PER_BLOCK + wave;
  const int nWaves  = gridDim.x * WAVES_PER_BLOCK;

  for (int t = waveGid; t < nTiles; t += nWaves) {
    const int nodeBase = t << 4;
    {
      const int r = lane & 15;
      const int node0 = nodeBase + r;
      const bool valid = node0 < N;
      const int node = valid ? node0 : 0;
      _Float16* row = buf0 + r * U1K;
      if (lane < 16) {
        const float4* hp = (const float4*)(hIn + (size_t)node * 16);
#pragma unroll
        for (int j = 0; j < 4; ++j) {
          float4 v = hp[j];
          if (!valid) { v.x = 0.f; v.y = 0.f; v.z = 0.f; v.w = 0.f; }
          row[j * 4 + 0] = (_Float16)v.x; row[j * 4 + 1] = (_Float16)v.y;
          row[j * 4 + 2] = (_Float16)v.z; row[j * 4 + 3] = (_Float16)v.w;
        }
      } else {
        const float dval = valid ? fmaxf(deg[node], 1.f) : 1.f;
        const float inv = 1.f / dval;
        const float4* ap = (const float4*)(aggr + (size_t)node * 16);
#pragma unroll
        for (int j = 0; j < 4; ++j) {
          float4 v = ap[j];
          if (!valid) { v.x = 0.f; v.y = 0.f; v.z = 0.f; v.w = 0.f; }
          row[16 + j * 4 + 0] = (_Float16)(v.x * inv);
          row[16 + j * 4 + 1] = (_Float16)(v.y * inv);
          row[16 + j * 4 + 2] = (_Float16)(v.z * inv);
          row[16 + j * 4 + 3] = (_Float16)(v.w * inv);
        }
      }
    }
    mlp_layer(buf0, buf1, sW1, b1, U1K, U1N, 70, lane, true);
    v8f acc = mlp_layer_acc(buf1, sW2, U2K, lane);
    {
      const int n = lane & 15;
      const int moff = (lane & 16) ? 8 : 0;
      const float bv = b2[n];
#pragma unroll
      for (int r = 0; r < 8; ++r) {
        const int node = nodeBase + r + moff;
        if (node < N) hOut[(size_t)node * 16 + n] = acc[r] + bv;
      }
    }
  }
}

// ---------------- final MLP + scatter-mean pool ----------------
__global__ __launch_bounds__(TPB) void final_kernel(
    const float* __restrict__ h, const int* __restrict__ batch, const float* __restrict__ cnt,
    const _Float16* __restrict__ gw1, const _Float16* __restrict__ gw2, const _Float16* __restrict__ gw3,
    const float* __restrict__ b1, const float* __restrict__ b2, const float* __restrict__ b3,
    float* __restrict__ out, int N)
{
  __shared__ __align__(16) _Float16 sW1[F1N * F1K];
  __shared__ __align__(16) _Float16 sW2[F2N * F2K];
  __shared__ __align__(16) _Float16 sW3[16 * F3K];
  __shared__ __align__(16) _Float16 sBuf0[WAVES_PER_BLOCK][16 * F1K];
  __shared__ __align__(16) _Float16 sBuf1[WAVES_PER_BLOCK][16 * F2K];

  const int tid = threadIdx.x, lane = tid & 31, wave = tid >> 5;
  copy_lds_async(gw1, sW1, F1N * F1K, tid);
  copy_lds_async(gw2, sW2, F2N * F2K, tid);
  copy_lds_async(gw3, sW3, 16 * F3K, tid);
  async_wait_all();
  __syncthreads();

  _Float16* buf0 = sBuf0[wave];
  _Float16* buf1 = sBuf1[wave];

  const int nTiles = (N + 15) >> 4;
  const int waveGid = blockIdx.x * WAVES_PER_BLOCK + wave;
  const int nWaves  = gridDim.x * WAVES_PER_BLOCK;

  for (int t = waveGid; t < nTiles; t += nWaves) {
    const int nodeBase = t << 4;
    {
      const int r = lane & 15;
      const int node0 = nodeBase + r;
      const bool valid = node0 < N;
      const int node = valid ? node0 : 0;
      _Float16* row = buf0 + r * F1K;
      if (lane < 16) {
        const float4* hp = (const float4*)(h + (size_t)node * 16);
#pragma unroll
        for (int j = 0; j < 4; ++j) {
          float4 v = hp[j];
          if (!valid) { v.x = 0.f; v.y = 0.f; v.z = 0.f; v.w = 0.f; }
          row[j * 4 + 0] = (_Float16)v.x; row[j * 4 + 1] = (_Float16)v.y;
          row[j * 4 + 2] = (_Float16)v.z; row[j * 4 + 3] = (_Float16)v.w;
        }
      } else {
#pragma unroll
        for (int c = 0; c < 16; ++c) row[16 + c] = (_Float16)0.f;  // K-pad 16..31
      }
    }
    mlp_layer(buf0, buf1, sW1, b1, F1K, F1N, 64, lane, true);
    mlp_layer(buf1, buf0, sW2, b2, F2K, F2N, 32, lane, true);
    v8f acc = mlp_layer_acc(buf0, sW3, F3K, lane);
    {
      const int n = lane & 15;
      const int moff = (lane & 16) ? 8 : 0;
      if (n < 3) {
        const float bv = b3[n];
#pragma unroll
        for (int r = 0; r < 8; ++r) {
          const int node = nodeBase + r + moff;
          if (node < N) {
            const int bg = batch[node];
            const float c = fmaxf(cnt[bg], 1.f);
            atomicAdd(&out[bg * 3 + n], (acc[r] + bv) / c);
          }
        }
      }
    }
  }
}

// ---------------- host orchestration ----------------
extern "C" void kernel_launch(void* const* d_in, const int* in_sizes, int n_in,
                              void* d_out, int out_size, void* d_ws, size_t ws_size,
                              hipStream_t stream) {
  constexpr int N = 100000, E = 3200000, B = 64, H = 16;
  (void)n_in; (void)ws_size; (void)B;

  const float* x     = (const float*)d_in[0];
  const int*   ei    = (const int*)d_in[1];
  const float* eattr = (const float*)d_in[2];
  const int*   batch = (const int*)d_in[3];
  const int* srcIdx = ei;       // edge_index[0]
  const int* dstIdx = ei + E;   // edge_index[1]

  // pytree leaf order: enc(W,b), final(3x W,b), layers(3x [msg 4x W,b; upd 2x W,b])
  // detect whether "final" precedes "layers" (jax sorted-dict order) via sizes.
  int finBase, layBase;
  if (in_sizes[6] == 16 * 64) { finBase = 6; layBase = 12; }
  else                        { layBase = 6; finBase = 6 + 36; }

  char* ws = (char*)d_ws;
  size_t off = 0;
  auto carve = [&](size_t bytes) -> void* {
    void* p = ws + off;
    off = (off + bytes + 255) & ~(size_t)255;
    return p;
  };
  float* h0   = (float*)carve(sizeof(float) * (size_t)N * H);
  float* h1   = (float*)carve(sizeof(float) * (size_t)N * H);
  float* aggr = (float*)carve(sizeof(float) * (size_t)N * H);
  float* deg  = (float*)carve(sizeof(float) * N);
  float* cnt  = (float*)carve(sizeof(float) * 64);

  // {din, dout, Kpad, Npad}
  const int msgDims[4][4] = {{35,70,M1K,M1N},{70,140,M2K,M2N},{140,20,M3K,M3N},{20,16,M4K,16}};
  const int updDims[2][4] = {{32,70,U1K,U1N},{70,16,U2K,16}};
  const int finDims[3][4] = {{16,64,F1K,F1N},{64,32,F2K,F2N},{32,3,F3K,16}};

  _Float16* msgW[3][4]; _Float16* updW[3][2]; _Float16* finW[3];
  const float* msgB[3][4]; const float* updB[3][2]; const float* finB[3];
  for (int l = 0; l < 3; ++l) {
    const int base = layBase + 12 * l;
    for (int j = 0; j < 4; ++j) {
      msgW[l][j] = (_Float16*)carve(sizeof(_Float16) * msgDims[j][2] * msgDims[j][3]);
      msgB[l][j] = (const float*)d_in[base + 2 * j + 1];
    }
    for (int j = 0; j < 2; ++j) {
      updW[l][j] = (_Float16*)carve(sizeof(_Float16) * updDims[j][2] * updDims[j][3]);
      updB[l][j] = (const float*)d_in[base + 8 + 2 * j + 1];
    }
  }
  for (int j = 0; j < 3; ++j) {
    finW[j] = (_Float16*)carve(sizeof(_Float16) * finDims[j][2] * finDims[j][3]);
    finB[j] = (const float*)d_in[finBase + 2 * j + 1];
  }

  auto zero = [&](float* p, int n) {
    zero_kernel<<<(n + 255) / 256, 256, 0, stream>>>(p, n);
  };
  zero(deg, N);
  zero(cnt, 64);
  zero((float*)d_out, out_size);

  auto prep = [&](const float* src, _Float16* dst, const int* d) {
    const int tot = d[2] * d[3];
    prep_w_kernel<<<(tot + 255) / 256, 256, 0, stream>>>(src, dst, d[0], d[1], d[2], d[3]);
  };
  for (int l = 0; l < 3; ++l) {
    const int base = layBase + 12 * l;
    for (int j = 0; j < 4; ++j) prep((const float*)d_in[base + 2 * j], msgW[l][j], msgDims[j]);
    for (int j = 0; j < 2; ++j) prep((const float*)d_in[base + 8 + 2 * j], updW[l][j], updDims[j]);
  }
  for (int j = 0; j < 3; ++j) prep((const float*)d_in[finBase + 2 * j], finW[j], finDims[j]);

  degree_kernel<<<(E + 255) / 256, 256, 0, stream>>>(dstIdx, deg, E);
  count_kernel<<<(N + 255) / 256, 256, 0, stream>>>(batch, cnt, N);

  encoder_kernel<<<((N * 16) + 255) / 256, 256, 0, stream>>>(
      x, (const float*)d_in[4], (const float*)d_in[5], h0, N);

  float* hc = h0; float* hn = h1;
  for (int l = 0; l < 3; ++l) {
    zero(aggr, N * H);
    edge_msg_kernel<<<512, TPB, 0, stream>>>(hc, srcIdx, dstIdx, eattr,
        msgW[l][0], msgW[l][1], msgW[l][2], msgW[l][3],
        msgB[l][0], msgB[l][1], msgB[l][2], msgB[l][3], aggr, E);
    node_update_kernel<<<128, TPB, 0, stream>>>(hc, aggr, deg,
        updW[l][0], updW[l][1], updB[l][0], updB[l][1], hn, N);
    float* t = hc; hc = hn; hn = t;
  }
  final_kernel<<<80, TPB, 0, stream>>>(hc, batch, cnt,
      finW[0], finW[1], finW[2], finB[0], finB[1], finB[2], (float*)d_out, N);
}
